// GCN_6150393168336
// MI455X (gfx1250) — matrix operand order repaired and checked by reading, not compiled, not run on hardware
//
#include <hip/hip_runtime.h>

// ---------------------------------------------------------------------------
// Problem constants (match reference)
// ---------------------------------------------------------------------------
#define N_NODES   100000
#define N_EDGES   1000000
#define IN_F      64
#define HID       64
#define N_GRAPHS  1000
#define HEADS     4
#define NOTES     128
#define NUM_MPNNS 4
#define NUM_CONVS 2
#define EPS_BN    1e-5f

typedef __bf16 bf16;
typedef __attribute__((ext_vector_type(16))) __bf16 v16bf;
typedef __attribute__((ext_vector_type(8)))  float  v8f;

__device__ __forceinline__ unsigned f2bf_bits(float f) {
    unsigned u = __float_as_uint(f);
    u += 0x7FFFu + ((u >> 16) & 1u);          // round-to-nearest-even
    return u >> 16;                            // bf16 bits in [15:0]
}

__device__ __forceinline__ bf16 f2bf(float f) {
    unsigned short hs = (unsigned short)f2bf_bits(f);
    union { unsigned short s; bf16 b; } cvt;
    cvt.s = hs;
    return cvt.b;
}

union Frag { unsigned u[8]; v16bf v; };

// Fragment index helpers (CDNA5 ISA 7.12.2, wave32, bf16 16x16x32):
//   A pair word index within a 32-word row: kb*16 + (vv>>2)*8 + hi*4 + (vv&3)
//   B pair word index within a 32-word col: kb*16 + (vv>>2)*8 + hi*4 + (vv&3)
__device__ __forceinline__ int aIdx(int kb, int vv, int hi) {
    return kb * 16 + ((vv >> 2) * 8) + hi * 4 + (vv & 3);
}
__device__ __forceinline__ int bIdx(int kb, int vv, int hi) {
    return kb * 16 + ((vv >> 2) * 8) + hi * 4 + (vv & 3);
}

// ---------------------------------------------------------------------------
// Generic WMMA GEMM: out[M x NOUT] = A[M x 64] * W[64 x NOUT] (+bias,+relu)
//   Used for: projection (BN fused), pooled@Wo, logits.
// ---------------------------------------------------------------------------
template <int NOUT, int EPI, bool BN>
__global__ __launch_bounds__(256)
void gemm64_wmma(const float* __restrict__ A, const bf16* __restrict__ W,
                 const float* __restrict__ bias,
                 const float* __restrict__ bnScale, const float* __restrict__ bnShift,
                 float* __restrict__ out, int M)
{
    constexpr int COLT = NOUT / 16;
    constexpr int ROWS = (COLT == 4) ? 32 : 16;

    __shared__ unsigned sA32[ROWS * 32];
    __shared__ unsigned sWt32[NOUT * 32];

    const int tid     = threadIdx.x;
    const int rowBase = blockIdx.x * ROWS;

    const unsigned short* Wu = (const unsigned short*)W;
#pragma unroll
    for (int i = tid; i < NOUT * 32; i += 256) {
        const int n  = i >> 5;
        const int kp = i & 31;
        unsigned lo = Wu[(2 * kp) * NOUT + n];
        unsigned hi = Wu[(2 * kp + 1) * NOUT + n];
        sWt32[n * 32 + kp] = lo | (hi << 16);
    }
#pragma unroll
    for (int i = tid; i < ROWS * 32; i += 256) {
        const int r  = i >> 5;
        const int cp = i & 31;
        const int g  = rowBase + r;
        float v0 = 0.0f, v1 = 0.0f;
        if (g < M) {
            const float* ap = A + (unsigned)g * 64 + 2 * cp;
            v0 = ap[0];
            v1 = ap[1];
        }
        if (BN) {
            v0 = v0 * bnScale[2 * cp]     + bnShift[2 * cp];
            v1 = v1 * bnScale[2 * cp + 1] + bnShift[2 * cp + 1];
        }
        sA32[r * 32 + cp] = f2bf_bits(v0) | (f2bf_bits(v1) << 16);
    }
    __syncthreads();

    const int wave = tid >> 5;
    const int lane = tid & 31;
    const int rt   = (COLT == 4) ? (wave >> 2) : 0;
    const int ct   = (COLT == 4) ? (wave & 3)  : wave;
    const int rowOff = rt * 16;
    const int mrow   = lane & 15;
    const int hi     = lane >> 4;

    v8f c = {0.f, 0.f, 0.f, 0.f, 0.f, 0.f, 0.f, 0.f};
    const unsigned* aRow = sA32 + (rowOff + mrow) * 32;
    const unsigned* bCol = sWt32 + (ct * 16 + mrow) * 32;
#pragma unroll
    for (int kb = 0; kb < 2; ++kb) {
        Frag fa, fb;
#pragma unroll
        for (int vv = 0; vv < 8; ++vv) {
            fa.u[vv] = aRow[aIdx(kb, vv, hi)];
            fb.u[vv] = bCol[bIdx(kb, vv, hi)];
        }
        c = __builtin_amdgcn_wmma_f32_16x16x32_bf16(false, fa.v, false, fb.v,
                                                    (short)0, c, false, false);
    }

    const int   ocol  = ct * 16 + mrow;
    const float bv    = (EPI >= 1) ? bias[ocol] : 0.0f;
    const int   orow0 = rowBase + rowOff + hi * 8;
    float* po = out + (unsigned)orow0 * NOUT + ocol;

    if (orow0 + 8 <= M) {
#pragma unroll
        for (int i = 0; i < 8; ++i) {
            float val = c[i] + bv;
            if (EPI == 2) val = fmaxf(val, 0.0f);
            po[i * NOUT] = val;
        }
    } else {
#pragma unroll
        for (int i = 0; i < 8; ++i) {
            if (orow0 + i < M) {
                float val = c[i] + bv;
                if (EPI == 2) val = fmaxf(val, 0.0f);
                po[i * NOUT] = val;
            }
        }
    }
}

// ---------------------------------------------------------------------------
// Fused GIN MLP: out = L2(relu(L1(relu(L0(agg)))))  (all 64x64, M%32==0)
//   One global read of agg, one global write of z. Intermediate activations
//   ping-pong through LDS as bf16 (they are the next layer's A fragments).
//   Also accumulates per-column sum / sum-of-squares for the following
//   BatchNorm (LDS reduce + global atomics), replacing a full stats pass.
// ---------------------------------------------------------------------------
__global__ __launch_bounds__(256)
void fused_mlp3(const float* __restrict__ agg,
                const bf16* __restrict__ W3,     // 3 consecutive 64x64 panels
                const float* __restrict__ b3,    // 3 consecutive 64 biases
                float* __restrict__ out,
                float* __restrict__ st_sum, float* __restrict__ st_sumsq)
{
    __shared__ unsigned       sWt32[3 * 64 * 32];    // 24KB transposed packed W
    __shared__ unsigned short act[2][32 * 64];       // 8KB bf16 act ping-pong
    __shared__ float          redS[64], redS2[64];

    const int tid     = threadIdx.x;
    const int rowBase = blockIdx.x * 32;

    const unsigned short* Wu = (const unsigned short*)W3;
#pragma unroll
    for (int i = tid; i < 3 * 64 * 32; i += 256) {
        const int p  = i >> 11;         // panel
        const int n  = (i >> 5) & 63;   // output column
        const int kp = i & 31;          // K pair
        unsigned lo = Wu[p * 4096 + (2 * kp) * 64 + n];
        unsigned hi = Wu[p * 4096 + (2 * kp + 1) * 64 + n];
        sWt32[i] = lo | (hi << 16);
    }
#pragma unroll
    for (int i = tid; i < 32 * 64; i += 256)
        act[0][i] = (unsigned short)f2bf_bits(agg[(unsigned)rowBase * 64 + i]);
    if (tid < 64) { redS[tid] = 0.f; redS2[tid] = 0.f; }
    __syncthreads();

    const int wave = tid >> 5, lane = tid & 31;
    const int rt = wave >> 2, ct = wave & 3;
    const int mrow = lane & 15, hi = lane >> 4;
    const int ocol = ct * 16 + mrow;
    const int rowOff = rt * 16;
    const int orow0 = rowOff + hi * 8;          // block-local row of 1st elem

    float keep[8];
#pragma unroll
    for (int l = 0; l < 3; ++l) {
        const unsigned* aRow = (const unsigned*)act[l & 1] + (rowOff + mrow) * 32;
        const unsigned* bCol = sWt32 + l * 2048 + ocol * 32;
        v8f c = {0.f, 0.f, 0.f, 0.f, 0.f, 0.f, 0.f, 0.f};
#pragma unroll
        for (int kb = 0; kb < 2; ++kb) {
            Frag fa, fb;
#pragma unroll
            for (int vv = 0; vv < 8; ++vv) {
                fa.u[vv] = aRow[aIdx(kb, vv, hi)];
                fb.u[vv] = bCol[bIdx(kb, vv, hi)];
            }
            c = __builtin_amdgcn_wmma_f32_16x16x32_bf16(false, fa.v, false, fb.v,
                                                        (short)0, c, false, false);
        }
        const float bv = b3[l * 64 + ocol];
        if (l < 2) {
            unsigned short* w = act[(l + 1) & 1] + orow0 * 64 + ocol;
#pragma unroll
            for (int i = 0; i < 8; ++i) {
                float val = fmaxf(c[i] + bv, 0.0f);
                w[i * 64] = (unsigned short)f2bf_bits(val);
            }
            __syncthreads();
        } else {
#pragma unroll
            for (int i = 0; i < 8; ++i) keep[i] = c[i] + bv;
        }
    }

    // write final layer + feed BatchNorm statistics
    float s = 0.f, s2 = 0.f;
    float* po = out + (unsigned)(rowBase + orow0) * 64 + ocol;
#pragma unroll
    for (int i = 0; i < 8; ++i) {
        float val = keep[i];
        po[i * 64] = val;
        s += val; s2 += val * val;
    }
    atomicAdd(&redS[ocol], s);
    atomicAdd(&redS2[ocol], s2);
    __syncthreads();
    if (tid < 64) {
        atomicAdd(&st_sum[tid],   redS[tid]);
        atomicAdd(&st_sumsq[tid], redS2[tid]);
    }
}

// ---------------------------------------------------------------------------
// Fused attention front-end: k = h@Wk (kept in LDS only), v = h@Wv -> global,
// scores = (k . seed)/sqrt(dh) -> global. A fragment shared by k/v WMMAs.
// M % 32 == 0.
// ---------------------------------------------------------------------------
__global__ __launch_bounds__(256)
void fused_kv_scores(const float* __restrict__ h, const bf16* __restrict__ Wk,
                     const bf16* __restrict__ Wv, const float* __restrict__ seed,
                     float* __restrict__ sc, float* __restrict__ vOut)
{
    __shared__ unsigned       sWtK[64 * 32];
    __shared__ unsigned       sWtV[64 * 32];
    __shared__ unsigned short sA[32 * 64];
    __shared__ float          sK[32 * 64];

    const int tid     = threadIdx.x;
    const int rowBase = blockIdx.x * 32;

    const unsigned short* WuK = (const unsigned short*)Wk;
    const unsigned short* WuV = (const unsigned short*)Wv;
#pragma unroll
    for (int i = tid; i < 64 * 32; i += 256) {
        const int n  = i >> 5;
        const int kp = i & 31;
        sWtK[i] = (unsigned)WuK[(2 * kp) * 64 + n] |
                  ((unsigned)WuK[(2 * kp + 1) * 64 + n] << 16);
        sWtV[i] = (unsigned)WuV[(2 * kp) * 64 + n] |
                  ((unsigned)WuV[(2 * kp + 1) * 64 + n] << 16);
    }
#pragma unroll
    for (int i = tid; i < 32 * 64; i += 256)
        sA[i] = (unsigned short)f2bf_bits(h[(unsigned)rowBase * 64 + i]);
    __syncthreads();

    const int wave = tid >> 5, lane = tid & 31;
    const int rt = wave >> 2, ct = wave & 3;
    const int mrow = lane & 15, hi = lane >> 4;
    const int ocol = ct * 16 + mrow;
    const int rowOff = rt * 16;
    const int orow0 = rowOff + hi * 8;

    const unsigned* aRow  = (const unsigned*)sA + (rowOff + mrow) * 32;
    const unsigned* bColK = sWtK + ocol * 32;
    const unsigned* bColV = sWtV + ocol * 32;

    v8f ck = {0.f, 0.f, 0.f, 0.f, 0.f, 0.f, 0.f, 0.f};
    v8f cv = {0.f, 0.f, 0.f, 0.f, 0.f, 0.f, 0.f, 0.f};
#pragma unroll
    for (int kb = 0; kb < 2; ++kb) {
        Frag fa, fbk, fbv;
#pragma unroll
        for (int vv = 0; vv < 8; ++vv) {
            fa.u[vv]  = aRow[aIdx(kb, vv, hi)];
            fbk.u[vv] = bColK[bIdx(kb, vv, hi)];
            fbv.u[vv] = bColV[bIdx(kb, vv, hi)];
        }
        ck = __builtin_amdgcn_wmma_f32_16x16x32_bf16(false, fa.v, false, fbk.v,
                                                     (short)0, ck, false, false);
        cv = __builtin_amdgcn_wmma_f32_16x16x32_bf16(false, fa.v, false, fbv.v,
                                                     (short)0, cv, false, false);
    }

    float* po = vOut + (unsigned)(rowBase + orow0) * 64 + ocol;
#pragma unroll
    for (int i = 0; i < 8; ++i) {
        sK[(orow0 + i) * 64 + ocol] = ck[i];
        po[i * 64] = cv[i];
    }
    __syncthreads();

    if (tid < 32 * HEADS) {
        const int n  = tid >> 2;
        const int hd = tid & 3;
        float s = 0.f;
#pragma unroll
        for (int d = 0; d < 16; ++d)
            s += sK[n * 64 + hd * 16 + d] * seed[hd * 16 + d];
        sc[(unsigned)(rowBase + n) * HEADS + hd] = s * 0.25f;   // 1/sqrt(16)
    }
}

// ---------------------------------------------------------------------------
// Small helper kernels
// ---------------------------------------------------------------------------
__global__ void cvt_bf16_kernel(const float* __restrict__ src,
                                bf16* __restrict__ dst, int n)
{
    int i = blockIdx.x * 256 + threadIdx.x;
    if (i < n) dst[i] = f2bf(src[i]);
}

__global__ __launch_bounds__(256)
void col_stats(const float* __restrict__ Z, int M,
               float* __restrict__ sum, float* __restrict__ sumsq)
{
    __shared__ float s1[256], s2[256];
    const int f  = threadIdx.x & 63;
    const int rl = threadIdx.x >> 6;
    float a = 0.f, a2 = 0.f;
    for (int r = blockIdx.x * 4 + rl; r < M; r += gridDim.x * 4) {
        float x = Z[(unsigned)r * 64 + f];
        a += x; a2 += x * x;
    }
    s1[threadIdx.x] = a; s2[threadIdx.x] = a2;
    __syncthreads();
    if (rl == 0) {
        atomicAdd(&sum[f],   s1[f] + s1[64 + f] + s1[128 + f] + s1[192 + f]);
        atomicAdd(&sumsq[f], s2[f] + s2[64 + f] + s2[128 + f] + s2[192 + f]);
    }
}

__global__ void finalize_stats(const float* __restrict__ sum,
                               const float* __restrict__ sumsq,
                               const float* __restrict__ gamma,
                               const float* __restrict__ beta,
                               float* __restrict__ scale,
                               float* __restrict__ shift, float invM)
{
    int f = threadIdx.x;    // 64 threads
    float mu  = sum[f] * invM;
    float var = sumsq[f] * invM - mu * mu;
    float s   = gamma[f] * rsqrtf(var + EPS_BN);
    scale[f] = s;
    shift[f] = beta[f] - mu * s;
}

// BN + relu; optionally duplicates the result into `seed` (next conv's agg
// self-term), eliminating a separate 25.6MB device-to-device copy.
__global__ void bn_relu_apply(const float* __restrict__ z,
                              const float* __restrict__ scale,
                              const float* __restrict__ shift,
                              float* __restrict__ out,
                              float* __restrict__ seed, int n)
{
    int i = blockIdx.x * 256 + threadIdx.x;
    if (i < n) {
        int f = i & 63;
        float v = fmaxf(z[i] * scale[f] + shift[f], 0.0f);
        out[i] = v;
        if (seed) seed[i] = v;
    }
}

// h += res; res = h; optionally seed = h (next conv's agg self-term)
__global__ void residual_update(float* __restrict__ h, float* __restrict__ res,
                                float* __restrict__ seed, int n)
{
    int i = blockIdx.x * 256 + threadIdx.x;
    if (i < n) {
        float v = h[i] + res[i];
        h[i] = v;
        res[i] = v;
        if (seed) seed[i] = v;
    }
}

// res = h; seed = h   (initial residual + first conv's agg self-term)
__global__ void seed_res(const float* __restrict__ h, float* __restrict__ res,
                         float* __restrict__ seed, int n)
{
    int i = blockIdx.x * 256 + threadIdx.x;
    if (i < n) {
        float v = h[i];
        res[i]  = v;
        seed[i] = v;
    }
}

__global__ __launch_bounds__(256)
void scatter_add(const float* __restrict__ h, const int* __restrict__ src,
                 const int* __restrict__ dst, float* __restrict__ agg, int nE)
{
    int t = blockIdx.x * 256 + threadIdx.x;
    int e = t >> 2;
    if (e >= nE) return;
    int q = t & 3;
    const float4* hp = (const float4*)(h + (unsigned)src[e] * 64 + q * 16);
    float* ap = agg + (unsigned)dst[e] * 64 + q * 16;
#pragma unroll
    for (int i = 0; i < 4; ++i) {
        float4 v = hp[i];
        atomicAdd(ap + 4 * i + 0, v.x);
        atomicAdd(ap + 4 * i + 1, v.y);
        atomicAdd(ap + 4 * i + 2, v.z);
        atomicAdd(ap + 4 * i + 3, v.w);
    }
}

__global__ void fill_neg_inf(float* __restrict__ p, int n)
{
    int i = blockIdx.x * 256 + threadIdx.x;
    if (i < n) p[i] = __int_as_float(0xff800000);
}

__global__ void seg_max(const float* __restrict__ sc, const int* __restrict__ batch,
                        float* __restrict__ gmax, int N)
{
    int t = blockIdx.x * 256 + threadIdx.x;
    if (t >= N * HEADS) return;
    int n = t >> 2, hd = t & 3;
    float v = sc[t];
    float* addr = &gmax[(unsigned)batch[n] * HEADS + hd];
    if (v >= 0.f) atomicMax((int*)addr, __float_as_int(v));
    else          atomicMin((unsigned int*)addr, __float_as_uint(v));
}

__global__ void seg_expsum(float* __restrict__ sc, const int* __restrict__ batch,
                           const float* __restrict__ gmax, float* __restrict__ gsum, int N)
{
    int t = blockIdx.x * 256 + threadIdx.x;
    if (t >= N * HEADS) return;
    int n = t >> 2, hd = t & 3;
    int g = batch[n];
    float e = __expf(sc[t] - gmax[(unsigned)g * HEADS + hd]);
    sc[t] = e;
    atomicAdd(&gsum[(unsigned)g * HEADS + hd], e);
}

__global__ void seg_pool(const float* __restrict__ sc, const float* __restrict__ gsum,
                         const float* __restrict__ V, const int* __restrict__ batch,
                         float* __restrict__ pooled, int N)
{
    int t = blockIdx.x * 256 + threadIdx.x;
    if (t >= N * 64) return;
    int n = t >> 6, j = t & 63, hd = j >> 4;
    int g = batch[n];
    float a = sc[(unsigned)n * HEADS + hd] / gsum[(unsigned)g * HEADS + hd];
    atomicAdd(&pooled[(unsigned)g * 64 + j], a * V[(unsigned)n * 64 + j]);
}

// ---------------------------------------------------------------------------
// Host orchestration
// ---------------------------------------------------------------------------
extern "C" void kernel_launch(void* const* d_in, const int* in_sizes, int n_in,
                              void* d_out, int out_size, void* d_ws, size_t ws_size,
                              hipStream_t stream)
{
    (void)in_sizes; (void)n_in; (void)out_size; (void)ws_size;

    const float* x          = (const float*)d_in[0];
    const int*   edge       = (const int*)d_in[1];
    const int*   src        = edge;
    const int*   dst        = edge + N_EDGES;
    const int*   batch      = (const int*)d_in[2];
    const float* fn_gamma   = (const float*)d_in[3];
    const float* fn_beta    = (const float*)d_in[4];
    const float* proj_W     = (const float*)d_in[5];
    const float* proj_b     = (const float*)d_in[6];
    const float* mlp_W      = (const float*)d_in[7];
    const float* mlp_b      = (const float*)d_in[8];
    const float* norm_gamma = (const float*)d_in[9];
    const float* norm_beta  = (const float*)d_in[10];
    const float* attn_seed  = (const float*)d_in[11];
    const float* attn_Wk    = (const float*)d_in[12];
    const float* attn_Wv    = (const float*)d_in[13];
    const float* attn_Wo    = (const float*)d_in[14];
    const float* pred_W     = (const float*)d_in[15];
    const float* pred_b     = (const float*)d_in[16];

    const size_t NB = (size_t)N_NODES * 64 * sizeof(float);
    char* ws = (char*)d_ws;
    size_t off = 0;
    auto alloc = [&](size_t bytes) -> void* {
        void* p = ws + off;
        off = (off + bytes + 255) & ~(size_t)255;
        return p;
    };
    float* h      = (float*)alloc(NB);
    float* res    = (float*)alloc(NB);
    float* t0     = (float*)alloc(NB);     // agg seed / agg
    float* t1     = (float*)alloc(NB);     // MLP output z / v
    float* sc     = (float*)alloc((size_t)N_NODES * HEADS * sizeof(float));
    float* gmax   = (float*)alloc((size_t)N_GRAPHS * HEADS * sizeof(float));
    float* gsum   = (float*)alloc((size_t)N_GRAPHS * HEADS * sizeof(float));
    float* pooled = (float*)alloc((size_t)N_GRAPHS * 64 * sizeof(float));
    float* stats  = (float*)alloc(256 * sizeof(float));   // sum|sumsq|scale|shift
    float* st_sum   = stats;
    float* st_sumsq = stats + 64;
    float* bnScale  = stats + 128;
    float* bnShift  = stats + 192;
    bf16* wbProj = (bf16*)alloc(64 * 64 * sizeof(bf16));
    bf16* wbMlp  = (bf16*)alloc(NUM_MPNNS * 3 * 64 * 64 * sizeof(bf16));
    bf16* wbK    = (bf16*)alloc(64 * 64 * sizeof(bf16));
    bf16* wbV    = (bf16*)alloc(64 * 64 * sizeof(bf16));
    bf16* wbO    = (bf16*)alloc(64 * 64 * sizeof(bf16));
    bf16* wbPred = (bf16*)alloc(64 * NOTES * sizeof(bf16));

    const int EW   = N_NODES * 64;
    const int EWB  = (EW + 255) / 256;
    const int GB32 = N_NODES / 32;               // 100000 % 32 == 0
    auto cvt = [&](const float* s, bf16* d, int n) {
        cvt_bf16_kernel<<<(n + 255) / 256, 256, 0, stream>>>(s, d, n);
    };

    // --- weights -> bf16 ---
    cvt(proj_W, wbProj, 64 * 64);
    cvt(mlp_W,  wbMlp,  NUM_MPNNS * 3 * 64 * 64);
    cvt(attn_Wk, wbK, 64 * 64);
    cvt(attn_Wv, wbV, 64 * 64);
    cvt(attn_Wo, wbO, 64 * 64);
    cvt(pred_W, wbPred, 64 * NOTES);

    // --- feature BatchNorm stats over x; apply fused into projection GEMM ---
    hipMemsetAsync(stats, 0, 128 * sizeof(float), stream);
    col_stats<<<512, 256, 0, stream>>>(x, N_NODES, st_sum, st_sumsq);
    finalize_stats<<<1, 64, 0, stream>>>(st_sum, st_sumsq, fn_gamma, fn_beta,
                                         bnScale, bnShift, 1.0f / N_NODES);

    // h = relu(BN(x) @ proj_W + proj_b)
    gemm64_wmma<64, 2, true><<<GB32, 256, 0, stream>>>(x, wbProj, proj_b,
                                                       bnScale, bnShift, h, N_NODES);
    // res = h; t0 = h (first conv's agg self-term)
    seed_res<<<EWB, 256, 0, stream>>>(h, res, t0, EW);

    // --- GIN blocks ---
    const int scatterBlocks = (4 * N_EDGES + 255) / 256;
    for (int m = 0; m < NUM_MPNNS; ++m) {
        for (int cv = 0; cv < NUM_CONVS; ++cv) {
            // agg (t0, pre-seeded with h) += scatter_sum(h[src] -> dst)
            scatter_add<<<scatterBlocks, 256, 0, stream>>>(h, src, dst, t0, N_EDGES);

            // z = MLP(agg)  (3 layers fused, BN stats accumulated in epilogue)
            hipMemsetAsync(stats, 0, 128 * sizeof(float), stream);
            fused_mlp3<<<GB32, 256, 0, stream>>>(t0, wbMlp + m * 3 * 4096,
                                                 mlp_b + m * 3 * 64, t1,
                                                 st_sum, st_sumsq);
            finalize_stats<<<1, 64, 0, stream>>>(st_sum, st_sumsq,
                                                 norm_gamma + m * 64, norm_beta + m * 64,
                                                 bnScale, bnShift, 1.0f / N_NODES);
            // h = relu(BN(z)); conv 0 also seeds next conv's agg
            bn_relu_apply<<<EWB, 256, 0, stream>>>(t1, bnScale, bnShift, h,
                                                   (cv == 0) ? t0 : nullptr, EW);
        }
        // h += res; res = h; seed next block's agg (except after last block)
        residual_update<<<EWB, 256, 0, stream>>>(h, res,
                                                 (m < NUM_MPNNS - 1) ? t0 : nullptr, EW);
    }

    // --- attention readout: k (LDS-only) -> scores, v -> t1 ---
    fused_kv_scores<<<GB32, 256, 0, stream>>>(h, wbK, wbV, attn_seed, sc, t1);

    const int NH = N_NODES * HEADS;
    fill_neg_inf<<<(N_GRAPHS * HEADS + 255) / 256, 256, 0, stream>>>(gmax, N_GRAPHS * HEADS);
    hipMemsetAsync(gsum, 0, (size_t)N_GRAPHS * HEADS * sizeof(float), stream);
    hipMemsetAsync(pooled, 0, (size_t)N_GRAPHS * 64 * sizeof(float), stream);
    seg_max<<<(NH + 255) / 256, 256, 0, stream>>>(sc, batch, gmax, N_NODES);
    seg_expsum<<<(NH + 255) / 256, 256, 0, stream>>>(sc, batch, gmax, gsum, N_NODES);
    seg_pool<<<EWB, 256, 0, stream>>>(sc, gsum, t1, batch, pooled, N_NODES);

    // embed = pooled @ Wo ; logits = embed @ pred_W + pred_b
    float* embed  = (float*)d_out;                       // [1000 x 64]
    float* logits = embed + (size_t)N_GRAPHS * 64;       // [1000 x 128]
    gemm64_wmma<64, 0, false><<<(N_GRAPHS + 31) / 32, 256, 0, stream>>>(
        pooled, wbO, nullptr, nullptr, nullptr, embed, N_GRAPHS);
    gemm64_wmma<128, 1, false><<<(N_GRAPHS + 15) / 16, 256, 0, stream>>>(
        embed, wbPred, pred_b, nullptr, nullptr, logits, N_GRAPHS);
}
